// PixelRNN_63333587747408
// MI455X (gfx1250) — matrix operand order repaired and checked by reading, not compiled, not run on hardware
//
#include <hip/hip_runtime.h>
#include <hip/hip_bf16.h>
#include <cmath>

// ---------------------------------------------------------------------------
// PixelRNN forward for MI455X (gfx1250, wave32, WMMA).
//
// Diagonal-LSTM recurrence (7 layers x 127 steps) is the serial bottleneck;
// per-step matmul work is a 64x32 @ 32x2048 f16 GEMM mapped to
// v_wmma_f32_16x16x32_f16.  One workgroup per (batch, direction) keeps the
// recurrence WGP-local (h packed f16 in LDS, c in registers in the WMMA C/D
// lane layout): only workgroup barriers, never grid syncs.
//
// Critical-path design:
//  * layer input kept as an f16 NHWC shadow (xT) so one diagonal column is
//    two global_load_b128 per lane, branch-free (clamped addr + AND mask),
//    issued under the WMMAs of the previous step
//  * LDS h state ping-pong double buffered -> ONE barrier per step
//  * branch-free B-operand build (selects, no divergent LDS clauses)
//  * branch-free sigmoid/tanh via v_exp_f32 + v_rcp_f32
//  * NHWC fwd/bwd outputs -> two float4 stores per lane per step
// ---------------------------------------------------------------------------

typedef __attribute__((ext_vector_type(16))) _Float16 v16h;
typedef __attribute__((ext_vector_type(8)))  float    v8f;
typedef __attribute__((ext_vector_type(2)))  __fp16   v2fp;   // cvt_pkrtz ret

union U16H { v16h v; _Float16 h[16]; unsigned u[8]; };
union U8F  { v8f  v; float    f[8];  };
union UPK  { v2fp h2; unsigned u; };

__device__ __forceinline__ float fast_sigmoid(float x) {
    float e = __expf(-x);                       // v_exp_f32
    return __builtin_amdgcn_rcpf(1.0f + e);     // v_rcp_f32
}

__device__ __forceinline__ float fast_tanh(float x) {
    float ax = fabsf(x);
    float e  = __expf(-2.0f * ax);
    float r  = __builtin_amdgcn_rcpf(1.0f + e);
    return copysignf((1.0f - e) * r, x);
}

__device__ __forceinline__ unsigned pack2h(float a, float b) {
    UPK t;
    t.h2 = __builtin_amdgcn_cvt_pkrtz(a, b);    // v_cvt_pk_rtz_f16_f32
    return t.u;
}

// ---------------------------------------------------------------------------
// Masked 7x7 conv, 1 -> 16 ch (rows 0..2 full, row 3 cols 0..2).  Writes the
// canonical NCHW f32 state AND the f16 NHWC shadow used by the LSTM kernel.
// ---------------------------------------------------------------------------
__global__ void conv_in_kernel(const float* __restrict__ X,
                               const float* __restrict__ W,
                               const float* __restrict__ Bv,
                               float* __restrict__ H,
                               _Float16* __restrict__ xT) {
    int idx = blockIdx.x * blockDim.x + threadIdx.x;
    if (idx >= 32 * 16 * 64 * 64) return;
    int x = idx & 63;
    int y = (idx >> 6) & 63;
    int o = (idx >> 12) & 15;
    int b = idx >> 16;
    float acc = Bv[o];
    #pragma unroll
    for (int r = 0; r < 4; ++r) {
        int iy = y + r - 3;                 // iy <= y < 64 always
        if (iy < 0) continue;
        int cmax = (r < 3) ? 7 : 3;
        for (int cc = 0; cc < cmax; ++cc) {
            int ix = x + cc - 3;
            if (ix < 0 || ix >= 64) continue;
            acc += W[o * 49 + r * 7 + cc] * X[(b * 64 + iy) * 64 + ix];
        }
    }
    H[idx] = acc;
    xT[(((b * 64 + y) * 64 + x) << 4) + o] = (_Float16)acc;
}

// ---------------------------------------------------------------------------
// One diagonal LSTM pass.  Grid (32 batches, 2 dirs); block 128 = 4 waves;
// wave w owns H rows [16w, 16w+16).  Per step t, per gate-tile mt (o,f,i,g):
//   acc = WMMA16x16x32( A1=[wi|ws1], B1=[x_col ; h], C=bias )
//   acc = WMMA16x16x32( A2=[ws0|0 ], B2=[h_up  ; 0], C=acc  )
// Layouts used here, contiguous K (A: lane%16=M, lanes<16 K=0..15, lanes>=16 K=16..31;
// B: lane%16=N, lanes<16 K=0..15, lanes>=16 K=16..31; C/D: lane%16=N,
// VGPR j = M=j (+8 upper)).
// ---------------------------------------------------------------------------
__global__ __launch_bounds__(128)
void bilstm_diag_kernel(const _Float16* __restrict__ xT,   // (B,H,W,16) f16
                        const float* __restrict__ wi_base,
                        const float* __restrict__ bi_base,
                        const float* __restrict__ ws0_base,
                        const float* __restrict__ ws1_base,
                        float* __restrict__ fwd_t,         // (B,H,W,16) f32
                        float* __restrict__ bwd_t) {
    const int b    = blockIdx.x;
    const int dir  = blockIdx.y;           // 0 = forward, 1 = reversed-W
    const int lane = threadIdx.x & 31;
    const int wv   = threadIdx.x >> 5;

    const float* wi  = wi_base  + dir * 64 * 16;
    const float* bi  = bi_base  + dir * 64;
    const float* ws0 = ws0_base + dir * 64 * 16;
    const float* ws1 = ws1_base + dir * 64 * 16;
    float* outp = dir ? bwd_t : fwd_t;

    // h state: ping-pong, 64 rows x 16 ch packed as 8x u32 (half2) per row.
    __shared__ unsigned hl32[2][64][8];
    for (int i = threadIdx.x; i < 64 * 8; i += blockDim.x)
        ((unsigned*)hl32[0])[i] = 0u;

    // ---- pack loop-invariant A operands (weights) into WMMA A layout ----
    U16H a1[4], a2[4];
    const int mrow = lane & 15;
    #pragma unroll
    for (int mt = 0; mt < 4; ++mt) {
        int oc = mt * 16 + mrow;           // gate output channel
        #pragma unroll
        for (int e = 0; e < 16; ++e) {
            int k = e;
            if (lane >= 16) k += 16;
            float va = (k < 16) ? wi[oc * 16 + k]  : ws1[oc * 16 + (k - 16)];
            float vb = (k < 16) ? ws0[oc * 16 + k] : 0.0f;
            a1[mt].h[e] = (_Float16)va;
            a2[mt].h[e] = (_Float16)vb;
        }
    }

    // ---- bias preloaded in the C/D lane layout ----
    const int chbase = (lane >= 16) ? 8 : 0;
    U8F biasv[4];
    #pragma unroll
    for (int mt = 0; mt < 4; ++mt)
        #pragma unroll
        for (int j = 0; j < 8; ++j)
            biasv[mt].f[j] = bi[mt * 16 + chbase + j];

    float creg[8];                          // cell state, C/D layout
    #pragma unroll
    for (int j = 0; j < 8; ++j) creg[j] = 0.0f;

    const int n   = lane & 15;              // N column within tile
    const int y   = wv * 16 + n;            // global H row
    const bool lo = (lane < 16);
    const _Float16* xrow = xT + (((b * 64 + y) * 64) << 4);  // this lane's row

    // Branch-free x-column fetch: clamp w into range, load b128 x2, AND-mask.
    auto fetch_xcol = [&](int t) -> U16H {
        U16H r;
        int w0 = t - y;
        unsigned m = (w0 >= 0 && w0 < 64) ? 0xFFFFFFFFu : 0u;
        int w0c = min(max(w0, 0), 63);
        int wx  = dir ? (63 - w0c) : w0c;
        const uint4* xp = (const uint4*)(xrow + (wx << 4));
        uint4 q0 = xp[0], q1 = xp[1];
        r.u[0] = q0.x & m; r.u[1] = q0.y & m; r.u[2] = q0.z & m; r.u[3] = q0.w & m;
        r.u[4] = q1.x & m; r.u[5] = q1.y & m; r.u[6] = q1.z & m; r.u[7] = q1.w & m;
        return r;
    };

    U16H xcur = fetch_xcol(0);
    __syncthreads();

    for (int t = 0; t < 127; ++t) {
        const int p  = t & 1;
        const int pn = p ^ 1;

        // ---- one LDS h-row per lane: y-1 for low half-wave, y for high ----
        U16H hrow;
        {
            int rowidx = lo ? ((y > 0) ? (y - 1) : 0) : y;
            const uint4* hp = (const uint4*)(&hl32[p][rowidx][0]);
            uint4 q0 = hp[0], q1 = hp[1];
            hrow.u[0] = q0.x; hrow.u[1] = q0.y; hrow.u[2] = q0.z; hrow.u[3] = q0.w;
            hrow.u[4] = q1.x; hrow.u[5] = q1.y; hrow.u[6] = q1.z; hrow.u[7] = q1.w;
        }

        // ---- branch-free B-operand build ----
        U16H b1, b2;
        const unsigned m2 = (lo && y > 0) ? 0xFFFFFFFFu : 0u;
        #pragma unroll
        for (int q = 0; q < 8; ++q) {
            b1.u[q] = lo ? xcur.u[q] : hrow.u[q];   // [x_col ; h]
            b2.u[q] = hrow.u[q] & m2;               // [h_up  ; 0]
        }

        // ---- gates: two chained WMMAs per 16-channel gate tile ----
        U8F acc[4];
        #pragma unroll
        for (int mt = 0; mt < 4; ++mt) {
            v8f t0 = __builtin_amdgcn_wmma_f32_16x16x32_f16(
                false, a1[mt].v, false, b1.v, (short)0, biasv[mt].v,
                false, false);
            acc[mt].v = __builtin_amdgcn_wmma_f32_16x16x32_f16(
                false, a2[mt].v, false, b2.v, (short)0, t0,
                false, false);
        }

        // ---- next diagonal's x column: issued under WMMAs/gate math ----
        U16H xnext = fetch_xcol(t + 1);

        // ---- per-lane LSTM gate math (o,f,i,g = tiles 0..3), branch-free --
        float hreg[8];
        #pragma unroll
        for (int j = 0; j < 8; ++j) {
            float og = fast_sigmoid(acc[0].f[j]);
            float fg = fast_sigmoid(acc[1].f[j]);
            float ig = fast_sigmoid(acc[2].f[j]);
            float gg = fast_tanh(acc[3].f[j]);
            float cn = fg * creg[j] + ig * gg;
            creg[j] = cn;
            hreg[j] = og * fast_tanh(cn);
        }

        // ---- new h -> ping-pong LDS buffer as packed half2 words ----
        #pragma unroll
        for (int j2 = 0; j2 < 4; ++j2)
            hl32[pn][y][(chbase >> 1) + j2] = pack2h(hreg[2 * j2],
                                                     hreg[2 * j2 + 1]);

        // ---- scatter to NHWC output (unskew): two float4 stores ----
        int w0 = t - y;
        if (w0 >= 0 && w0 < 64) {
            int wx = dir ? (63 - w0) : w0;
            float4* op = (float4*)(outp + ((((b * 64 + y) * 64) + wx) << 4)
                                   + chbase);
            op[0] = make_float4(hreg[0], hreg[1], hreg[2], hreg[3]);
            op[1] = make_float4(hreg[4], hreg[5], hreg[6], hreg[7]);
        }

        __syncthreads();        // single barrier: pn writes visible
        xcur = xnext;
    }
}

// ---------------------------------------------------------------------------
// h(NCHW) = h + fwd + shift_down_H(bwd); also refresh the f16 NHWC shadow.
// ---------------------------------------------------------------------------
__global__ void combine_kernel(float* __restrict__ h,
                               const float* __restrict__ fwd_t,
                               const float* __restrict__ bwd_t,
                               _Float16* __restrict__ xT) {
    int idx = blockIdx.x * blockDim.x + threadIdx.x;
    if (idx >= 32 * 16 * 64 * 64) return;
    int w = idx & 63;
    int y = (idx >> 6) & 63;
    int c = (idx >> 12) & 15;
    int b = idx >> 16;
    int tix = (((b * 64 + y) * 64 + w) << 4) + c;
    float bv = (y > 0) ? bwd_t[tix - (64 << 4)] : 0.0f;
    float v  = h[idx] + fwd_t[tix] + bv;
    h[idx]  = v;
    xT[tix] = (_Float16)v;
}

// ---------------------------------------------------------------------------
// Output head: relu(W1 h + b1) -> relu(W2 . + b2) -> W3 . + b3.
// Tiny K=16 matmuls, memory bound; weight indices uniform -> scalar loads.
// ---------------------------------------------------------------------------
__global__ void head_kernel(const float* __restrict__ h,
                            const float* __restrict__ w1, const float* __restrict__ b1,
                            const float* __restrict__ w2, const float* __restrict__ b2,
                            const float* __restrict__ w3, const float* __restrict__ b3,
                            float* __restrict__ out) {
    int p = blockIdx.x * blockDim.x + threadIdx.x;   // pixel over B*H*W
    if (p >= 32 * 64 * 64) return;
    int b  = p >> 12;
    int yx = p & 4095;
    float v[16], t1[16], t2[16];
    #pragma unroll
    for (int c = 0; c < 16; ++c) v[c] = h[(b * 16 + c) * 4096 + yx];
    #pragma unroll
    for (int o = 0; o < 16; ++o) {
        float a = b1[o];
        #pragma unroll
        for (int c = 0; c < 16; ++c) a += w1[o * 16 + c] * v[c];
        t1[o] = fmaxf(a, 0.0f);
    }
    #pragma unroll
    for (int o = 0; o < 16; ++o) {
        float a = b2[o];
        #pragma unroll
        for (int c = 0; c < 16; ++c) a += w2[o * 16 + c] * t1[c];
        t2[o] = fmaxf(a, 0.0f);
    }
    float a = b3[0];
    #pragma unroll
    for (int c = 0; c < 16; ++c) a += w3[c] * t2[c];
    out[p] = a;
}

// ---------------------------------------------------------------------------
extern "C" void kernel_launch(void* const* d_in, const int* in_sizes, int n_in,
                              void* d_out, int out_size, void* d_ws, size_t ws_size,
                              hipStream_t stream) {
    (void)in_sizes; (void)n_in; (void)out_size; (void)ws_size;

    const float* X      = (const float*)d_in[0];
    const float* conv_w = (const float*)d_in[1];
    const float* conv_b = (const float*)d_in[2];
    const float* i2s_w  = (const float*)d_in[3];   // (7,2,64,16)
    const float* i2s_b  = (const float*)d_in[4];   // (7,2,64)
    const float* ss_w0  = (const float*)d_in[5];   // (7,2,64,16)
    const float* ss_w1  = (const float*)d_in[6];   // (7,2,64,16)
    const float* ow1    = (const float*)d_in[7];
    const float* ob1    = (const float*)d_in[8];
    const float* ow2    = (const float*)d_in[9];
    const float* ob2    = (const float*)d_in[10];
    const float* ow3    = (const float*)d_in[11];
    const float* ob3    = (const float*)d_in[12];

    const int NELT = 32 * 16 * 64 * 64;            // 2 Mi elements
    float*    h     = (float*)d_ws;                // NCHW f32, 8 MB
    float*    fwd_t = h + NELT;                    // NHWC f32, 8 MB
    float*    bwd_t = fwd_t + NELT;                // NHWC f32, 8 MB
    _Float16* xT    = (_Float16*)(bwd_t + NELT);   // NHWC f16, 4 MB

    conv_in_kernel<<<NELT / 256, 256, 0, stream>>>(X, conv_w, conv_b, h, xT);

    for (int l = 0; l < 7; ++l) {
        bilstm_diag_kernel<<<dim3(32, 2), 128, 0, stream>>>(
            xT,
            i2s_w + l * 2 * 64 * 16,
            i2s_b + l * 2 * 64,
            ss_w0 + l * 2 * 64 * 16,
            ss_w1 + l * 2 * 64 * 16,
            fwd_t, bwd_t);
        combine_kernel<<<NELT / 256, 256, 0, stream>>>(h, fwd_t, bwd_t, xT);
    }

    head_kernel<<<(32 * 64 * 64) / 256, 256, 0, stream>>>(
        h, ow1, ob1, ow2, ob2, ow3, ob3, (float*)d_out);
}
